// ManualRNNDecryptor_35923106464032
// MI455X (gfx1250) — matrix operand order repaired and checked by reading, not compiled, not run on hardware
//
#include <hip/hip_runtime.h>
#include <stdint.h>

// ---------------------------------------------------------------------------
// Types for CDNA5 WMMA (wave32): 16x16x32 bf16 -> f32
// ---------------------------------------------------------------------------
typedef __attribute__((ext_vector_type(16))) __bf16 bf16x16;
typedef __attribute__((ext_vector_type(8)))  float  f32x8;
typedef __attribute__((ext_vector_type(4)))  unsigned int u32x4;
typedef __attribute__((ext_vector_type(4)))  int i32x4;
typedef __attribute__((ext_vector_type(8)))  int i32x8;

union V16 {
    bf16x16 v;
    u32x4   q[2];
};

static __device__ __forceinline__ f32x8 wmma_bf16(bf16x16 a, bf16x16 b, f32x8 c) {
    // (neg_a, A, neg_b, B, c_mod, C, reuse_a, reuse_b)
    return __builtin_amdgcn_wmma_f32_16x16x32_bf16(false, a, false, b, (short)0, c,
                                                   false, false);
}

static __device__ __forceinline__ unsigned short f32_to_bf16(float f) {
    union { float f; unsigned int u; } v; v.f = f;
    unsigned int r = v.u + 0x7FFFu + ((v.u >> 16) & 1u);   // round-to-nearest-even
    return (unsigned short)(r >> 16);
}

static __device__ __forceinline__ float bf16_to_f32(unsigned short h) {
    union { unsigned int u; float f; } v; v.u = ((unsigned int)h) << 16;
    return v.f;
}

// Fast tanh on the gfx1250 transcendental unit. Prefer the native V_TANH op;
// else 1 - 2/(e^{2x}+1) with v_exp_f32 / v_rcp_f32 (saturates correctly to +-1).
static __device__ __forceinline__ float fast_tanh(float x) {
#if __has_builtin(__builtin_amdgcn_tanhf)
    return __builtin_amdgcn_tanhf(x);
#else
    const float t = __builtin_amdgcn_exp2f(x * 2.8853900817779268f);  // 2*log2(e)
    return 1.0f - 2.0f * __builtin_amdgcn_rcpf(t + 1.0f);
#endif
}

// ---------------------------------------------------------------------------
// CDNA5 Tensor Data Mover (TDM): 1-D bf16 tile DMA, descriptor per ISA 8.3-8.4
// ---------------------------------------------------------------------------
#if __has_builtin(__builtin_amdgcn_tensor_load_to_lds) && \
    __has_builtin(__builtin_amdgcn_tensor_store_from_lds) && \
    __has_builtin(__builtin_amdgcn_s_wait_tensorcnt)
#define HAVE_TDM 1
#if __clang_major__ >= 23
#define TDM_6ARG 1      // amdgpu-toolchain (therock HIP headers): 6-arg builtin
#endif
#endif

#ifdef HAVE_TDM
// Build D# groups 0/1 for a contiguous 1-D transfer of nelem 2-byte elements.
static __device__ __forceinline__ void tdm_desc_1d(u32x4& g0, i32x8& g1,
                                                   unsigned long long gaddr,
                                                   unsigned lds_byte_addr,
                                                   unsigned nelem) {
    g0[0] = 1u;                                            // count=1, user descriptor
    g0[1] = lds_byte_addr;                                 // lds_addr (bytes)
    g0[2] = (unsigned)(gaddr & 0xFFFFFFFFull);             // global_addr[31:0]
    g0[3] = (unsigned)((gaddr >> 32) & 0x01FFFFFFull)      // global_addr[56:32]
            | (2u << 30);                                  // type=2 ("image")
    g1[0] = (int)(1u << 16);                               // data_size=1 (2 bytes)
    g1[1] = (int)((nelem & 0xFFFFu) << 16);                // tensor_dim0[15:0]
    g1[2] = (int)(((nelem >> 16) & 0xFFFFu) | (1u << 16)); // tensor_dim0[31:16], dim1=1
    g1[3] = (int)((nelem & 0xFFFFu) << 16);                // tile_dim0
    g1[4] = 1;                                             // tile_dim1 = 1
    g1[5] = (int)nelem;                                    // tensor_dim0_stride[31:0]
    g1[6] = 0;
    g1[7] = 0;
}

static __device__ __forceinline__ void tdm_load_1d(const void* gsrc,
                                                   unsigned lds_byte_addr,
                                                   unsigned nelem) {
    u32x4 g0; i32x8 g1;
    tdm_desc_1d(g0, g1, (unsigned long long)(size_t)gsrc, lds_byte_addr, nelem);
    i32x4 gz = {0, 0, 0, 0};
#ifdef TDM_6ARG
    i32x8 gz8 = {0, 0, 0, 0, 0, 0, 0, 0};
    __builtin_amdgcn_tensor_load_to_lds(g0, g1, gz, gz, gz8, 0);
#else
    __builtin_amdgcn_tensor_load_to_lds(g0, g1, gz, gz, 0);
#endif
}

static __device__ __forceinline__ void tdm_store_1d(void* gdst,
                                                    unsigned lds_byte_addr,
                                                    unsigned nelem) {
    u32x4 g0; i32x8 g1;
    tdm_desc_1d(g0, g1, (unsigned long long)(size_t)gdst, lds_byte_addr, nelem);
    i32x4 gz = {0, 0, 0, 0};
#ifdef TDM_6ARG
    i32x8 gz8 = {0, 0, 0, 0, 0, 0, 0, 0};
    __builtin_amdgcn_tensor_store_from_lds(g0, g1, gz, gz, gz8, 0);
#else
    __builtin_amdgcn_tensor_store_from_lds(g0, g1, gz, gz, 0);
#endif
}

static __device__ __forceinline__ void tdm_wait() {
    __builtin_amdgcn_s_wait_tensorcnt(0);
}
#endif  // HAVE_TDM

// A tile (16x32 bf16, M x K), row-major source with leading dim ld.
static __device__ __forceinline__ bf16x16 load_a_tile(const unsigned short* p, int ld,
                                                      int k, int lane) {
    const int lg = lane >> 4, row = lane & 15;
    const unsigned short* a = p + (size_t)row * ld + k + 8 * lg;
    V16 u;
    u.q[0] = *(const u32x4*)(a);
    u.q[1] = *(const u32x4*)(a + 16);
    return u.v;
}

// Nontemporal variant for single-pass global streams.
static __device__ __forceinline__ bf16x16 load_a_tile_nt(const unsigned short* p, int ld,
                                                         int k, int lane) {
    const int lg = lane >> 4, row = lane & 15;
    const unsigned short* a = p + (size_t)row * ld + k + 8 * lg;
    V16 u;
    u.q[0] = __builtin_nontemporal_load((const u32x4*)(a));
    u.q[1] = __builtin_nontemporal_load((const u32x4*)(a + 16));
    return u.v;
}

// B tile (32x16 bf16, K x N) where B(k,n) = W[n][k], W row-major [N][K] ld=K.
static __device__ __forceinline__ bf16x16 load_b_tile(const unsigned short* w, int ld,
                                                      int n, int k, int lane) {
    const int lg = lane >> 4, nn = lane & 15;
    const unsigned short* b = w + (size_t)(n + nn) * ld + k + 8 * lg;
    V16 u;
    u.q[0] = *(const u32x4*)(b);
    u.q[1] = *(const u32x4*)(b + 16);
    return u.v;
}

// ---------------------------------------------------------------------------
// Sizes
// ---------------------------------------------------------------------------
#define BATCH 128
#define TSTEPS 512
#define EMB 512
#define HID 1024
#define OUTV 256
#define ROWS (TSTEPS * BATCH)   // 65536
#define MTILE 32                // batch rows per recurrence workgroup

// LDS layout of recurrence kernel (ushort units):
//   [0, 2*MTILE*HID)                        h double buffer (128 KB)
//   [2*MTILE*HID, 4*MTILE*HID)              xp bf16 double staging (2x64 KB)
#define LDS_STAGE_US (2 * MTILE * HID)          // ushort offset of staging
#define REC_LDS_BYTES ((size_t)4 * MTILE * HID * sizeof(unsigned short)) // 256 KB

// ---------------------------------------------------------------------------
// fp32 -> bf16 weight conversion
// ---------------------------------------------------------------------------
__global__ void __launch_bounds__(256) bf16_convert_kernel(const float* __restrict__ s,
                                                           unsigned short* __restrict__ d,
                                                           int n) {
    int i = blockIdx.x * 256 + threadIdx.x;
    if (i < n) d[i] = f32_to_bf16(s[i]);
}

// ---------------------------------------------------------------------------
// embProj[v][h] = bi0[h] + sum_k emb[v][k] * wi0[h][k]   (256 x 1024, fp32)
// ---------------------------------------------------------------------------
__global__ void __launch_bounds__(256) embproj_kernel(const float* __restrict__ emb,
                                                      const float* __restrict__ wi0,
                                                      const float* __restrict__ bi0,
                                                      float* __restrict__ embP) {
    __shared__ float er[EMB];
    const int v = blockIdx.x;
    for (int i = threadIdx.x; i < EMB; i += 256) er[i] = emb[(size_t)v * EMB + i];
    __syncthreads();
    for (int h = threadIdx.x; h < HID; h += 256) {
        const float* wr = wi0 + (size_t)h * EMB;
        float s = bi0[h];
        for (int k = 0; k < EMB; ++k) s += er[k] * wr[k];
        embP[(size_t)v * HID + h] = s;
    }
}

// ---------------------------------------------------------------------------
// Recurrent layer (persistent over T).  4 workgroups x 32 batch rows.
// 1024 threads = 32 waves; wave w owns output columns [32w, 32w+32).
// Per k-step each wave runs a 2x2 WMMA tile grid. h double-buffered in LDS.
// Wh streamed bf16 from L2 every step (un-hoistable via per-step clobber).
// TDM path: wave 0 DMAs next step's xp slab into LDS staging during the WMMA
// loop (TENSOR_LOAD_TO_LDS) and publishes h via TENSOR_STORE_FROM_LDS.
// ---------------------------------------------------------------------------
__global__ void __launch_bounds__(1024) rnn_layer_kernel(
    const unsigned short* __restrict__ Wh,    // [HID][HID] bf16
    const float* __restrict__ embP,           // [256][HID] (layer0) or null
    const int* __restrict__ x,                // [BATCH][TSTEPS] (layer0) or null
    const unsigned short* __restrict__ xp,    // [TSTEPS][BATCH][HID] bf16 (layer1)
    unsigned short* __restrict__ hs,          // [TSTEPS][BATCH][HID] bf16 out
    int layer0) {
    extern __shared__ unsigned short hb[];
    const int tid = threadIdx.x;
    const int lane = tid & 31;
    const int w = tid >> 5;                   // 0..31
    const int lg = lane >> 4;
    const int ln = lane & 15;
    const int btile = blockIdx.x;             // 0..3
    const int n0 = w * 32;
#ifdef HAVE_TDM
    // Wave-relative LDS byte base of the dynamic allocation (ISA 10.2: generic
    // LDS addresses carry the LDS offset in addr[31:0]).
    const unsigned ldsbase = (unsigned)(size_t)hb;
#endif

    // zero h(0)
    {
        u32x4 z = {0u, 0u, 0u, 0u};
        u32x4* p = (u32x4*)hb;
        for (int i = tid; i < (2 * MTILE * HID) / 8; i += 1024) p[i] = z;
    }

#ifdef HAVE_TDM
    // Pre-stage xp(0) for layer 1.
    if (!layer0 && w == 0) {
        tdm_load_1d(xp + (size_t)btile * MTILE * HID,
                    ldsbase + (unsigned)(LDS_STAGE_US * 2), MTILE * HID);
        tdm_wait();
    }
#endif
    __syncthreads();

    for (int t = 0; t < TSTEPS; ++t) {
        // Stop LICM from hoisting the (t-invariant) Wh tile loads out of the
        // t-loop — hoisting 2 KB/lane forces scratch spills; re-reading from
        // the L2-resident 2 MB Wh is far cheaper.
        asm volatile("" ::: "memory");

        const unsigned short* hA = hb + (size_t)(t & 1) * MTILE * HID;
        unsigned short*       hN = hb + (size_t)((t + 1) & 1) * MTILE * HID;

#ifdef HAVE_TDM
        // Kick off async staging of xp(t+1) while WMMAs run (critical path hide).
        if (!layer0 && w == 0 && t + 1 < TSTEPS) {
            const unsigned short* nxt =
                xp + ((size_t)(t + 1) * BATCH + btile * MTILE) * HID;
            tdm_load_1d(nxt,
                        ldsbase + (unsigned)((LDS_STAGE_US +
                                              ((t + 1) & 1) * MTILE * HID) * 2),
                        MTILE * HID);
        }
#else
        if (!layer0 && t + 1 < TSTEPS) {
            const unsigned short* nxt =
                xp + ((size_t)(t + 1) * BATCH + btile * MTILE) * HID +
                (size_t)tid * 32;
            __builtin_prefetch(nxt, 0, 0);
        }
#endif

        f32x8 acc00 = {}, acc01 = {}, acc10 = {}, acc11 = {};
        #pragma unroll 4
        for (int k = 0; k < HID; k += 32) {
            bf16x16 a0 = load_a_tile(hA, HID, k, lane);
            bf16x16 a1 = load_a_tile(hA + 16 * HID, HID, k, lane);
            bf16x16 b0 = load_b_tile(Wh, HID, n0,      k, lane);
            bf16x16 b1 = load_b_tile(Wh, HID, n0 + 16, k, lane);
            acc00 = wmma_bf16(a0, b0, acc00);
            acc01 = wmma_bf16(a0, b1, acc01);
            acc10 = wmma_bf16(a1, b0, acc10);
            acc11 = wmma_bf16(a1, b1, acc11);
        }

        // epilogue: add input projection, tanh, publish new h
#ifdef HAVE_TDM
        const unsigned short* stage = hb + LDS_STAGE_US + (size_t)(t & 1) * MTILE * HID;
#endif
        #pragma unroll
        for (int m = 0; m < 2; ++m) {
            #pragma unroll
            for (int j = 0; j < 2; ++j) {
                f32x8 acc = m ? (j ? acc11 : acc10) : (j ? acc01 : acc00);
                const int col = n0 + j * 16 + ln;
                #pragma unroll
                for (int i = 0; i < 8; ++i) {
                    const int r  = m * 16 + i + 8 * lg;   // 0..31 within tile
                    const int bg = btile * MTILE + r;     // global batch row
                    float xv;
                    if (layer0) {
                        const int xi = x[(size_t)bg * TSTEPS + t];
                        xv = embP[(size_t)xi * HID + col];
                    } else {
#ifdef HAVE_TDM
                        xv = bf16_to_f32(stage[(size_t)r * HID + col]);
#else
                        xv = bf16_to_f32(xp[((size_t)t * BATCH + bg) * HID + col]);
#endif
                    }
                    const float hv = fast_tanh(acc[i] + xv);
                    hN[(size_t)r * HID + col] = f32_to_bf16(hv);
                }
            }
        }

#ifdef HAVE_TDM
        // Ensure xp(t+1) staging (and hs store of t-1) are complete before the
        // barrier that publishes them to all waves.
        if (w == 0) tdm_wait();
        __syncthreads();
        // DMA-publish hs[t] from hN; safe: hN isn't rewritten until epilogue of
        // t+2, which is ordered after wave 0's tdm_wait at the end of t+1.
        if (w == 0) {
            tdm_store_1d(hs + ((size_t)t * BATCH + btile * MTILE) * HID,
                         ldsbase + (unsigned)((((t + 1) & 1) * MTILE * HID) * 2),
                         MTILE * HID);
        }
#else
        __syncthreads();
        {
            u32x4* dst = (u32x4*)(hs + ((size_t)t * BATCH + btile * MTILE) * HID);
            const u32x4* src = (const u32x4*)hN;
            #pragma unroll
            for (int c = 0; c < 4; ++c) dst[tid + c * 1024] = src[tid + c * 1024];
        }
#endif
    }
    // Final TDM store is drained by S_ENDPGM's implicit wait-idle.
}

// ---------------------------------------------------------------------------
// xp1 = bf16(hs0 @ Wi1^T + bi1) :  [65536,1024] x [1024,1024] -> bf16
// grid (ROWS/64, HID/64), block 128 (4 waves), wave = one 16-row M strip.
// ---------------------------------------------------------------------------
__global__ void __launch_bounds__(128) xproj_gemm_kernel(
    const unsigned short* __restrict__ A,     // hs0 bf16 [ROWS][HID]
    const unsigned short* __restrict__ Wb,    // Wi1 bf16 [HID][HID]
    const float* __restrict__ bias,           // [HID]
    unsigned short* __restrict__ C) {         // xp1 bf16 [ROWS][HID]
    const int tid = threadIdx.x;
    const int lane = tid & 31;
    const int w = tid >> 5;
    const int lg = lane >> 4;
    const int ln = lane & 15;
    const int m0 = blockIdx.x * 64 + w * 16;
    const int nb = blockIdx.y * 64;

    f32x8 acc[4] = {{}, {}, {}, {}};
    for (int k = 0; k < HID; k += 32) {
        bf16x16 a = load_a_tile_nt(A + (size_t)m0 * HID, HID, k, lane);
        #pragma unroll
        for (int j = 0; j < 4; ++j) {
            bf16x16 b = load_b_tile(Wb, HID, nb + j * 16, k, lane);
            acc[j] = wmma_bf16(a, b, acc[j]);
        }
    }
    #pragma unroll
    for (int j = 0; j < 4; ++j) {
        const int col = nb + j * 16 + ln;
        const float bv = bias[col];
        #pragma unroll
        for (int i = 0; i < 8; ++i) {
            const int r = m0 + i + 8 * lg;
            __builtin_nontemporal_store(f32_to_bf16(acc[j][i] + bv),
                                        &C[(size_t)r * HID + col]);
        }
    }
}

// ---------------------------------------------------------------------------
// LayerNorm(hs1) @ proj_w^T + proj_b -> out [B][T][OUTV]
// block 512 (16 waves), 64 rows/WG.
// ---------------------------------------------------------------------------
__global__ void __launch_bounds__(512) ln_proj_kernel(
    const unsigned short* __restrict__ H,     // hs1 bf16 [ROWS][HID]
    const float* __restrict__ ln_g,
    const float* __restrict__ ln_b,
    const unsigned short* __restrict__ Pw,    // proj_w bf16 [OUTV][HID]
    const float* __restrict__ pb,             // [OUTV]
    float* __restrict__ out) {                // [BATCH][TSTEPS][OUTV]
    extern __shared__ unsigned short nm[];    // [64][HID] bf16 = 128 KB
    const int tid = threadIdx.x;
    const int lane = tid & 31;
    const int w = tid >> 5;                   // 0..15
    const int lg = lane >> 4;
    const int ln = lane & 15;
    const int m0 = blockIdx.x * 64;

    for (int rr = 0; rr < 4; ++rr) {
        const int r = w * 4 + rr;             // 0..63
        const unsigned short* hp = H + (size_t)(m0 + r) * HID + lane * 32;
        float s = 0.f, ss = 0.f;
        #pragma unroll
        for (int c = 0; c < 32; ++c) {
            const float v = bf16_to_f32(hp[c]);
            s += v; ss += v * v;
        }
        #pragma unroll
        for (int off = 16; off > 0; off >>= 1) {
            s  += __shfl_xor(s,  off, 32);
            ss += __shfl_xor(ss, off, 32);
        }
        const float mu  = s * (1.f / (float)HID);
        const float var = ss * (1.f / (float)HID) - mu * mu;
        const float rs  = rsqrtf(var + 1e-5f);
        unsigned short* np = nm + (size_t)r * HID + lane * 32;
        #pragma unroll
        for (int c = 0; c < 32; ++c) {
            const int col = lane * 32 + c;
            const float v = (bf16_to_f32(hp[c]) - mu) * rs * ln_g[col] + ln_b[col];
            np[c] = f32_to_bf16(v);
        }
    }
    __syncthreads();

    const int ms = (w & 3) * 16;
    const int nq = (w >> 2) * 64;
    f32x8 acc[4] = {{}, {}, {}, {}};
    for (int k = 0; k < HID; k += 32) {
        bf16x16 a = load_a_tile(nm + (size_t)ms * HID, HID, k, lane);
        #pragma unroll
        for (int j = 0; j < 4; ++j) {
            bf16x16 b = load_b_tile(Pw, HID, nq + j * 16, k, lane);
            acc[j] = wmma_bf16(a, b, acc[j]);
        }
    }
    #pragma unroll
    for (int j = 0; j < 4; ++j) {
        const int col = nq + j * 16 + ln;
        const float pbv = pb[col];
        #pragma unroll
        for (int i = 0; i < 8; ++i) {
            const int r = m0 + ms + i + 8 * lg;   // global row = t*BATCH + b
            const int t = r >> 7;
            const int b = r & 127;
            __builtin_nontemporal_store(acc[j][i] + pbv,
                &out[((size_t)b * TSTEPS + t) * OUTV + col]);
        }
    }
}

// ---------------------------------------------------------------------------
// Host-side launch
// ---------------------------------------------------------------------------
extern "C" void kernel_launch(void* const* d_in, const int* in_sizes, int n_in,
                              void* d_out, int out_size, void* d_ws, size_t ws_size,
                              hipStream_t stream) {
    const float* emb    = (const float*)d_in[0];
    const float* wi0    = (const float*)d_in[1];
    const float* bi0    = (const float*)d_in[2];
    const float* wh0    = (const float*)d_in[3];
    const float* wi1    = (const float*)d_in[4];
    const float* bi1    = (const float*)d_in[5];
    const float* wh1    = (const float*)d_in[6];
    const float* ln_g   = (const float*)d_in[7];
    const float* ln_b   = (const float*)d_in[8];
    const float* proj_w = (const float*)d_in[9];
    const float* proj_b = (const float*)d_in[10];
    const int*   x      = (const int*)d_in[11];
    float* out = (float*)d_out;

    unsigned char* ws = (unsigned char*)d_ws;
    size_t off = 0;
    auto carve = [&](size_t bytes) -> void* {
        void* p = ws + off;
        off += (bytes + 255) & ~(size_t)255;
        return p;
    };
    unsigned short* wh0b = (unsigned short*)carve((size_t)HID * HID * 2);
    unsigned short* wi1b = (unsigned short*)carve((size_t)HID * HID * 2);
    unsigned short* wh1b = (unsigned short*)carve((size_t)HID * HID * 2);
    unsigned short* pwb  = (unsigned short*)carve((size_t)OUTV * HID * 2);
    float*          embP = (float*)carve((size_t)256 * HID * 4);
    unsigned short* hs0  = (unsigned short*)carve((size_t)ROWS * HID * 2);
    unsigned short* xp1  = (unsigned short*)carve((size_t)ROWS * HID * 2);
    unsigned short* hs1  = hs0;   // hs0 fully consumed by xproj_gemm before layer 1

    const int nW = HID * HID;     // 1048576
    bf16_convert_kernel<<<(nW + 255) / 256, 256, 0, stream>>>(wh0, wh0b, nW);
    bf16_convert_kernel<<<(nW + 255) / 256, 256, 0, stream>>>(wi1, wi1b, nW);
    bf16_convert_kernel<<<(nW + 255) / 256, 256, 0, stream>>>(wh1, wh1b, nW);
    bf16_convert_kernel<<<(OUTV * HID + 255) / 256, 256, 0, stream>>>(proj_w, pwb,
                                                                     OUTV * HID);
    embproj_kernel<<<256, 256, 0, stream>>>(emb, wi0, bi0, embP);

    // Layer 0 recurrence (gathers from embProj via x)
    rnn_layer_kernel<<<BATCH / MTILE, 1024, REC_LDS_BYTES, stream>>>(
        wh0b, embP, x, nullptr, hs0, 1);

    // Layer 1 input projection (fully parallel GEMM over T*B rows)
    xproj_gemm_kernel<<<dim3(ROWS / 64, HID / 64), 128, 0, stream>>>(hs0, wi1b, bi1,
                                                                     xp1);

    // Layer 1 recurrence (TDM-staged xp)
    rnn_layer_kernel<<<BATCH / MTILE, 1024, REC_LDS_BYTES, stream>>>(
        wh1b, nullptr, nullptr, xp1, hs1, 0);

    // LayerNorm + output projection
    ln_proj_kernel<<<ROWS / 64, 512, 64 * HID * sizeof(unsigned short), stream>>>(
        hs1, ln_g, ln_b, pwb, proj_b, out);
}
